// QRNNLayer_46849503265294
// MI455X (gfx1250) — matrix operand order repaired and checked by reading, not compiled, not run on hardware
//
#include <hip/hip_runtime.h>
#include <cmath>

// Problem dims (fixed by the reference)
#define S_DIM 2048
#define B_DIM 16
#define D_DIM 1024   // K
#define H_DIM 1024
#define M_DIM (S_DIM * B_DIM)   // 32768 rows
#define N_DIM (3 * H_DIM)       // 3072 cols

typedef __attribute__((ext_vector_type(16))) __bf16 v16bf;
typedef __attribute__((ext_vector_type(8)))  __bf16 v8bf;
typedef __attribute__((ext_vector_type(8)))  float  v8f;
typedef __attribute__((ext_vector_type(4)))  float  v4f;

// ---- one-shot f32 -> bf16 conversion (emits v_cvt_pk_bf16_f32) -----------
__global__ __launch_bounds__(256)
void cvt_f32_to_bf16(const float* __restrict__ in, __bf16* __restrict__ out,
                     int n /* multiple of 8 */) {
    int i = (blockIdx.x * blockDim.x + threadIdx.x) * 8;
    if (i >= n) return;
    v4f a = *(const v4f*)(in + i);
    v4f b = *(const v4f*)(in + i + 4);
    v8bf o;
#pragma unroll
    for (int r = 0; r < 4; ++r) {
        o[r]     = (__bf16)a[r];
        o[4 + r] = (__bf16)b[r];
    }
    *(v8bf*)(out + i) = o;
}

// ---- bf16 fragment loaders (no conversion in the hot loop) ---------------

// A fragment, 16x32 bf16, contiguous k order (k = 16*half + s), same as B:
//   lane<16 : a[0..7]=K(k0+0..7),   a[8..15]=K(k0+8..15)
//   lane>=16: a[0..7]=K(k0+16..23), a[8..15]=K(k0+24..31)
// Each 8-element chunk is a contiguous 16-byte global_load_b128.
__device__ __forceinline__ v16bf load_frag_a(const __bf16* __restrict__ xrow,
                                             int k0, int half) {
    v8bf lo = *(const v8bf*)(xrow + k0 + half * 16);
    v8bf hi = *(const v8bf*)(xrow + k0 + half * 16 + 8);
    return __builtin_shufflevector(lo, hi,
        0, 1, 2, 3, 4, 5, 6, 7, 8, 9, 10, 11, 12, 13, 14, 15);
}

// B fragment, 32x16 bf16: lane holds column n = lane&15,
// K-range = k0 + (lane>>4)*16 + 0..15 -> one contiguous 32-byte run of Wbf.
// wrow already includes the n*D + half*16 offset.
__device__ __forceinline__ v16bf load_frag_b(const __bf16* __restrict__ wrow,
                                             int k0) {
    return *(const v16bf*)(wrow + k0);
}

__device__ __forceinline__ float sigmoidf_dev(float x) {
    return 1.0f / (1.0f + __expf(-x));
}

// ---- GEMM + bias + activation kernel -------------------------------------
// grid = (N_DIM/64, M_DIM/256), block = 256 (8 waves).
// Wave w computes rows [mbase, mbase+32) x cols [nbase, nbase+64).
__global__ __launch_bounds__(256)
void qrnn_gemm_act(const __bf16* __restrict__ X,
                   const __bf16* __restrict__ W,
                   const float* __restrict__ bias,
                   float* __restrict__ Y) {
    const int lane = threadIdx.x & 31;
    const int wave = threadIdx.x >> 5;
    const int half = lane >> 4;
    const int l16  = lane & 15;

    const int nbase = blockIdx.x * 64;
    const int mbase = blockIdx.y * 256 + wave * 32;

    v8f acc[2][4];
#pragma unroll
    for (int i = 0; i < 2; ++i)
#pragma unroll
        for (int j = 0; j < 4; ++j)
            acc[i][j] = {};

    const __bf16* arow0 = X + (size_t)(mbase + l16) * D_DIM;
    const __bf16* arow1 = X + (size_t)(mbase + 16 + l16) * D_DIM;
    const __bf16* wrow0 = W + (size_t)(nbase + 0  + l16) * D_DIM + half * 16;
    const __bf16* wrow1 = W + (size_t)(nbase + 16 + l16) * D_DIM + half * 16;
    const __bf16* wrow2 = W + (size_t)(nbase + 32 + l16) * D_DIM + half * 16;
    const __bf16* wrow3 = W + (size_t)(nbase + 48 + l16) * D_DIM + half * 16;

    for (int k0 = 0; k0 < D_DIM; k0 += 32) {
        // prefetch the A stream two K-steps ahead (emits global_prefetch_b8)
        if (k0 + 64 < D_DIM) {
            __builtin_prefetch(arow0 + k0 + 64, 0, 0);
            __builtin_prefetch(arow1 + k0 + 64, 0, 0);
        }
        v16bf a0 = load_frag_a(arow0, k0, half);
        v16bf a1 = load_frag_a(arow1, k0, half);

        v16bf b0 = load_frag_b(wrow0, k0);
        v16bf b1 = load_frag_b(wrow1, k0);
        v16bf b2 = load_frag_b(wrow2, k0);
        v16bf b3 = load_frag_b(wrow3, k0);

        acc[0][0] = __builtin_amdgcn_wmma_f32_16x16x32_bf16(false, a0, false, b0, (short)0, acc[0][0], false, false);
        acc[1][0] = __builtin_amdgcn_wmma_f32_16x16x32_bf16(false, a1, false, b0, (short)0, acc[1][0], false, false);
        acc[0][1] = __builtin_amdgcn_wmma_f32_16x16x32_bf16(false, a0, false, b1, (short)0, acc[0][1], false, false);
        acc[1][1] = __builtin_amdgcn_wmma_f32_16x16x32_bf16(false, a1, false, b1, (short)0, acc[1][1], false, false);
        acc[0][2] = __builtin_amdgcn_wmma_f32_16x16x32_bf16(false, a0, false, b2, (short)0, acc[0][2], false, false);
        acc[1][2] = __builtin_amdgcn_wmma_f32_16x16x32_bf16(false, a1, false, b2, (short)0, acc[1][2], false, false);
        acc[0][3] = __builtin_amdgcn_wmma_f32_16x16x32_bf16(false, a0, false, b3, (short)0, acc[0][3], false, false);
        acc[1][3] = __builtin_amdgcn_wmma_f32_16x16x32_bf16(false, a1, false, b3, (short)0, acc[1][3], false, false);
    }

    // Epilogue: bias + activation + store (C/D layout: VGPR r -> m = r + 8*half,
    // lane&15 -> n). nbase is 64-aligned and H%64==0, so chunk is uniform per tile.
    const int chunk = nbase / H_DIM;  // 0 = Z (tanh), 1 = F (sigmoid), 2 = O (sigmoid)
#pragma unroll
    for (int j = 0; j < 4; ++j) {
        const int n = nbase + 16 * j + l16;
        const float bv = bias[n];
#pragma unroll
        for (int i = 0; i < 2; ++i) {
            const int mrow = mbase + i * 16 + half * 8;
#pragma unroll
            for (int r = 0; r < 8; ++r) {
                float v = acc[i][j][r] + bv;
                v = (chunk == 0) ? tanhf(v) : sigmoidf_dev(v);
                Y[(size_t)(mrow + r) * N_DIM + n] = v;
            }
        }
    }
}

// ---- sequential ForgetMult scan + output gate ----------------------------
// One thread per (b,h) channel; coalesced across h.
__global__ __launch_bounds__(256)
void qrnn_scan(const float* __restrict__ Y,
               const float* __restrict__ hidden,
               float* __restrict__ Hout,
               float* __restrict__ Clast) {
    const int t = blockIdx.x * blockDim.x + threadIdx.x;  // t = b*H + h
    if (t >= B_DIM * H_DIM) return;
    const int b = t >> 10;         // /H_DIM
    const int h = t & (H_DIM - 1);

    float c = hidden[t];
    const float* base = Y + (size_t)b * N_DIM + h;
    float* outp = Hout + t;

    for (int s = 0; s < S_DIM; ++s) {
        const float* p = base + (size_t)s * (B_DIM * N_DIM);
        const float z = p[0];
        const float f = p[H_DIM];
        const float o = p[2 * H_DIM];
        c = f * c + (1.0f - f) * z;
        outp[(size_t)s * (B_DIM * H_DIM)] = o * c;
    }
    Clast[t] = c;
}

// ---- launch ---------------------------------------------------------------
extern "C" void kernel_launch(void* const* d_in, const int* in_sizes, int n_in,
                              void* d_out, int out_size, void* d_ws, size_t ws_size,
                              hipStream_t stream) {
    const float* X      = (const float*)d_in[0];  // [S,B,D]
    const float* hidden = (const float*)d_in[1];  // [B,H]
    const float* W      = (const float*)d_in[2];  // [3H,D]
    const float* bias   = (const float*)d_in[3];  // [3H]

    float* out = (float*)d_out;                   // Hout [S,B,H] ++ C[-1] [1,B,H]

    // Workspace layout: Y f32 | X bf16 | W bf16
    float*  Y   = (float*)d_ws;                                    // M*N f32
    __bf16* Xbf = (__bf16*)((char*)d_ws + (size_t)M_DIM * N_DIM * sizeof(float));
    __bf16* Wbf = Xbf + (size_t)M_DIM * D_DIM;

    const int nX = M_DIM * D_DIM;                 // 33,554,432 (mult of 8)
    const int nW = N_DIM * D_DIM;                 //  3,145,728 (mult of 8)
    cvt_f32_to_bf16<<<(nX / 8 + 255) / 256, 256, 0, stream>>>(X, Xbf, nX);
    cvt_f32_to_bf16<<<(nW / 8 + 255) / 256, 256, 0, stream>>>(W, Wbf, nW);

    dim3 gemm_grid(N_DIM / 64, M_DIM / 256);      // (48, 128)
    qrnn_gemm_act<<<gemm_grid, 256, 0, stream>>>(Xbf, Wbf, bias, Y);

    const int nchan = B_DIM * H_DIM;              // 16384
    qrnn_scan<<<(nchan + 255) / 256, 256, 0, stream>>>(
        Y, hidden, out, out + (size_t)S_DIM * B_DIM * H_DIM);
}
